// OldTaskAssignmentNet_60318520705374
// MI455X (gfx1250) — compile-verified, matrix operand order checked by reading, not run on hardware
//
#include <hip/hip_runtime.h>
#include <hip/hip_fp16.h>

// ---------------- problem constants (match reference) ----------------
#define NH     2
#define NC     32
#define NHC    64           // H*C
#define NB     128          // graphs
#define TPER   1024
#define NTASKS (NB*TPER)    // 131072
#define NDATA  65536
#define NE     1048576
#define NDEV   4
#define TASKD  12
#define DATAD  5
#define DEVD   12
#define EDGED  3
#define EMBD   76           // 12 + 32 + 32
#define INHEAD 200          // 2*76 + 48

typedef __attribute__((ext_vector_type(16))) _Float16 v16h;
typedef __attribute__((ext_vector_type(8)))  float    v8f;

// order-preserving float<->uint encoding for atomicMax-based segment max
__device__ __forceinline__ unsigned f2ord(float f) {
    unsigned u = __float_as_uint(f);
    return (u & 0x80000000u) ? ~u : (u | 0x80000000u);
}
__device__ __forceinline__ float ord2f(unsigned u) {
    unsigned b = (u & 0x80000000u) ? (u & 0x7fffffffu) : ~u;
    return __uint_as_float(b);
}

// ---------------- zero init ----------------
__global__ void k_zero_u32(unsigned* __restrict__ p, size_t n) {
    size_t i = (size_t)blockIdx.x * blockDim.x + threadIdx.x;
    if (i < n) p[i] = 0u;
}

// ---------------- pack W [K,64] (f32, K<=32 after pad) into WMMA B-layout f16 ----------------
// B tile is 32x16 (KxN) per 16-col block; lane L holds column n = nb*16 + (L%16),
// kbase = (L<16)?0:8 ; vgpr j pairs: j<4 -> K = kbase+2j, j>=4 -> K = 16+kbase+2(j-4)
__global__ void k_pack_w(const float* __restrict__ W, int Kd, _Float16* __restrict__ Wp) {
    int idx = blockIdx.x * blockDim.x + threadIdx.x;   // 4*32*8 = 1024 items
    if (idx >= 4 * 32 * 8) return;
    int j    = idx & 7;
    int lane = (idx >> 3) & 31;
    int nb   = idx >> 8;
    int kbase = (lane < 16) ? 0 : 8;
    int k0 = ((j < 4) ? 0 : 16) + kbase + 2 * (j & 3);
    int n  = nb * 16 + (lane & 15);
    float f0 = (k0     < Kd) ? W[(size_t)k0 * NHC + n]       : 0.f;
    float f1 = (k0 + 1 < Kd) ? W[(size_t)(k0 + 1) * NHC + n] : 0.f;
    Wp[(size_t)(nb * 32 + lane) * 16 + 2 * j]     = (_Float16)f0;
    Wp[(size_t)(nb * 32 + lane) * 16 + 2 * j + 1] = (_Float16)f1;
}

// ---------------- projection: Y[M,64] = X[M,Kd] @ W + bias via WMMA ----------------
// one wave per 16-row tile; 4 WMMAs (one per 16-col block)
__global__ void k_proj(const float* __restrict__ X, int M, int Kd,
                       const _Float16* __restrict__ Wp,
                       const float* __restrict__ bias,
                       float* __restrict__ Y) {
    int wave = (blockIdx.x * blockDim.x + threadIdx.x) >> 5;
    int lane = threadIdx.x & 31;
    int row0 = wave * 16;
    if (row0 >= M) return;                       // wave-uniform (M multiple of 128)

    int r     = row0 + (lane & 15);
    int kbase = (lane < 16) ? 0 : 8;
    const float* xrow = X + (size_t)r * Kd;

    v16h a;
#pragma unroll
    for (int j = 0; j < 8; ++j) {
        int k0 = ((j < 4) ? 0 : 16) + kbase + 2 * (j & 3);
        float f0 = (k0     < Kd) ? xrow[k0]     : 0.f;
        float f1 = (k0 + 1 < Kd) ? xrow[k0 + 1] : 0.f;
        a[2 * j]     = (_Float16)f0;
        a[2 * j + 1] = (_Float16)f1;
    }

    int mbase = (lane < 16) ? 0 : 8;
#pragma unroll
    for (int nb = 0; nb < 4; ++nb) {
        v16h b = *(const v16h*)(Wp + (size_t)(nb * 32 + lane) * 16);
        float bv = bias[nb * 16 + (lane & 15)];
        v8f c;
#pragma unroll
        for (int q = 0; q < 8; ++q) c[q] = bv;
        c = __builtin_amdgcn_wmma_f32_16x16x32_f16(false, a, false, b,
                                                   (short)0, c, false, false);
        int n = nb * 16 + (lane & 15);
#pragma unroll
        for (int q = 0; q < 8; ++q)
            Y[(size_t)(row0 + mbase + q) * NHC + n] = c[q];
    }
}

// ---------------- edge pass 1: attention scores + segment max ----------------
template <bool HAS_E>
__global__ void k_edge_score(const int* __restrict__ src, const int* __restrict__ dst,
                             const float* __restrict__ xl, const float* __restrict__ xr,
                             const float* __restrict__ eattr, const float* __restrict__ We,
                             const float* __restrict__ att,
                             float* __restrict__ score, unsigned* __restrict__ smax) {
    __shared__ float sWe[EDGED * NHC];
    __shared__ float sAtt[NHC];
    if (threadIdx.x < NHC) sAtt[threadIdx.x] = att[threadIdx.x];
    if (HAS_E)
        for (int i = threadIdx.x; i < EDGED * NHC; i += blockDim.x) sWe[i] = We[i];
    __syncthreads();

    int e = blockIdx.x * blockDim.x + threadIdx.x;
    if (e >= NE) return;
    int s = src[e], d = dst[e];

    float ea0 = 0.f, ea1 = 0.f, ea2 = 0.f;
    if (HAS_E) {
        ea0 = eattr[(size_t)e * 3 + 0];
        ea1 = eattr[(size_t)e * 3 + 1];
        ea2 = eattr[(size_t)e * 3 + 2];
    }

    const float4* xs = (const float4*)(xl + (size_t)s * NHC);
    const float4* xd = (const float4*)(xr + (size_t)d * NHC);
    float sc[2] = {0.f, 0.f};
#pragma unroll
    for (int q = 0; q < 16; ++q) {
        float4 av = xs[q];
        float4 bv = xd[q];
        float mm[4] = {av.x + bv.x, av.y + bv.y, av.z + bv.z, av.w + bv.w};
#pragma unroll
        for (int c4 = 0; c4 < 4; ++c4) {
            int i = q * 4 + c4;
            float m = mm[c4];
            if (HAS_E) m += ea0 * sWe[i] + ea1 * sWe[NHC + i] + ea2 * sWe[2 * NHC + i];
            m = (m >= 0.f) ? m : 0.2f * m;          // leaky 0.2
            sc[i >> 5] += m * sAtt[i];
        }
    }
    score[(size_t)e * 2 + 0] = sc[0];
    score[(size_t)e * 2 + 1] = sc[1];
    atomicMax(&smax[(size_t)d * 2 + 0], f2ord(sc[0]));
    atomicMax(&smax[(size_t)d * 2 + 1], f2ord(sc[1]));
}

// ---------------- edge pass 2: exp, denom, weighted scatter-add ----------------
__global__ void k_edge_accum(const int* __restrict__ src, const int* __restrict__ dst,
                             const float* __restrict__ xl, const float* __restrict__ score,
                             const unsigned* __restrict__ smax,
                             float* __restrict__ denom, float* __restrict__ acc) {
    int e = blockIdx.x * blockDim.x + threadIdx.x;
    if (e >= NE) return;
    int s = src[e], d = dst[e];
    float e0 = __expf(score[(size_t)e * 2 + 0] - ord2f(smax[(size_t)d * 2 + 0]));
    float e1 = __expf(score[(size_t)e * 2 + 1] - ord2f(smax[(size_t)d * 2 + 1]));
    atomicAdd(&denom[(size_t)d * 2 + 0], e0);
    atomicAdd(&denom[(size_t)d * 2 + 1], e1);
    const float* xs = xl + (size_t)s * NHC;
    float* ad = acc + (size_t)d * NHC;
#pragma unroll 8
    for (int i = 0; i < NC; ++i)  atomicAdd(&ad[i],      e0 * xs[i]);
#pragma unroll 8
    for (int i = 0; i < NC; ++i)  atomicAdd(&ad[NC + i], e1 * xs[NC + i]);
}

// ---------------- finalize conv: normalize, head-mean, residual, LN, leaky ----------------
__global__ void k_finalize(const float* __restrict__ acc, const float* __restrict__ denom,
                           const float* __restrict__ tasks_x,
                           const float* __restrict__ Wres, const float* __restrict__ bias,
                           const float* __restrict__ g, const float* __restrict__ bln,
                           float* __restrict__ emb, int off, int copy_x) {
    __shared__ float sW[TASKD * NC];
    __shared__ float sb[NC], sg[NC], sl[NC];
    for (int i = threadIdx.x; i < TASKD * NC; i += blockDim.x) sW[i] = Wres[i];
    if (threadIdx.x < NC) {
        sb[threadIdx.x] = bias[threadIdx.x];
        sg[threadIdx.x] = g[threadIdx.x];
        sl[threadIdx.x] = bln[threadIdx.x];
    }
    __syncthreads();

    int t = blockIdx.x * blockDim.x + threadIdx.x;
    if (t >= NTASKS) return;

    float x[TASKD];
#pragma unroll
    for (int k = 0; k < TASKD; ++k) x[k] = tasks_x[(size_t)t * TASKD + k];

    float d0 = denom[(size_t)t * 2 + 0] + 1e-16f;
    float d1 = denom[(size_t)t * 2 + 1] + 1e-16f;
    const float* ap = acc + (size_t)t * NHC;

    float v[NC];
    float mean = 0.f;
#pragma unroll
    for (int c = 0; c < NC; ++c) {
        float val = 0.5f * (ap[c] / d0 + ap[NC + c] / d1);   // mean over heads
        float r = sb[c];
#pragma unroll
        for (int k = 0; k < TASKD; ++k) r += x[k] * sW[k * NC + c];
        v[c] = val + r;
        mean += v[c];
    }
    mean *= (1.f / NC);
    float var = 0.f;
#pragma unroll
    for (int c = 0; c < NC; ++c) { float dd = v[c] - mean; var += dd * dd; }
    var *= (1.f / NC);
    float inv = rsqrtf(var + 1e-5f);

    float* ep = emb + (size_t)t * EMBD;
#pragma unroll
    for (int c = 0; c < NC; ++c) {
        float y = (v[c] - mean) * inv * sg[c] + sl[c];
        y = (y >= 0.f) ? y : 0.01f * y;
        ep[off + c] = y;
    }
    if (copy_x) {
#pragma unroll
        for (int k = 0; k < TASKD; ++k) ep[k] = x[k];
    }
}

// ---------------- per-graph mean pool (graphs are contiguous 1024-task blocks) ----------------
__global__ void k_pool(const float* __restrict__ emb, const float* __restrict__ counts,
                       float* __restrict__ glob) {
    int b = blockIdx.x;
    int d = threadIdx.x;
    if (d >= EMBD) return;
    const float* p = emb + (size_t)b * TPER * EMBD + d;
    float s = 0.f;
    for (int r = 0; r < TPER; ++r) s += p[(size_t)r * EMBD];
    float cnt = counts[b];
    cnt = (cnt > 1.f) ? cnt : 1.f;
    glob[(size_t)b * EMBD + d] = s / cnt;
}

// ---------------- head MLP: [B,200] -> LN(leaky) 32 -> 3 ----------------
__global__ void k_head(const float* __restrict__ emb, const float* __restrict__ glob,
                       const float* __restrict__ dev, const int* __restrict__ ptr,
                       const float* __restrict__ W1, const float* __restrict__ b1,
                       const float* __restrict__ g, const float* __restrict__ bl,
                       const float* __restrict__ W2, const float* __restrict__ b2,
                       float* __restrict__ out) {
    int b = blockIdx.x * blockDim.x + threadIdx.x;
    if (b >= NB) return;
    const float* cand = emb + (size_t)ptr[b] * EMBD;
    const float* gp   = glob + (size_t)b * EMBD;
    const float* dp   = dev + (size_t)b * (NDEV * DEVD);

    float h[NC];
#pragma unroll
    for (int c = 0; c < NC; ++c) h[c] = b1[c];
    for (int k = 0; k < INHEAD; ++k) {
        float xv = (k < EMBD) ? cand[k]
                 : (k < 2 * EMBD) ? gp[k - EMBD]
                 : dp[k - 2 * EMBD];
        const float* wr = W1 + (size_t)k * NC;
#pragma unroll
        for (int c = 0; c < NC; ++c) h[c] += xv * wr[c];
    }
    float mean = 0.f;
#pragma unroll
    for (int c = 0; c < NC; ++c) mean += h[c];
    mean *= (1.f / NC);
    float var = 0.f;
#pragma unroll
    for (int c = 0; c < NC; ++c) { float dd = h[c] - mean; var += dd * dd; }
    var *= (1.f / NC);
    float inv = rsqrtf(var + 1e-5f);
#pragma unroll
    for (int c = 0; c < NC; ++c) {
        float y = (h[c] - mean) * inv * g[c] + bl[c];
        h[c] = (y >= 0.f) ? y : 0.01f * y;
    }
#pragma unroll
    for (int j = 0; j < NDEV - 1; ++j) {
        float o = b2[j];
#pragma unroll
        for (int c = 0; c < NC; ++c) o += h[c] * W2[c * (NDEV - 1) + j];
        out[(size_t)b * (NDEV - 1) + j] = o;
    }
}

// ---------------- launcher ----------------
extern "C" void kernel_launch(void* const* d_in, const int* in_sizes, int n_in,
                              void* d_out, int out_size, void* d_ws, size_t ws_size,
                              hipStream_t stream) {
    (void)in_sizes; (void)n_in; (void)out_size; (void)ws_size;
    const float* tasks_x   = (const float*)d_in[0];
    const float* data_x    = (const float*)d_in[1];
    const float* devices_x = (const float*)d_in[2];
    const int*   ei_dt     = (const int*)d_in[3];
    const float* eattr     = (const float*)d_in[4];
    const int*   ei_tt     = (const int*)d_in[5];
    const int*   ptr       = (const int*)d_in[7];
    const float* counts    = (const float*)d_in[8];
    const float* c1_Wl   = (const float*)d_in[9];
    const float* c1_bl   = (const float*)d_in[10];
    const float* c1_Wr   = (const float*)d_in[11];
    const float* c1_br   = (const float*)d_in[12];
    const float* c1_We   = (const float*)d_in[13];
    const float* c1_att  = (const float*)d_in[14];
    const float* c1_Wres = (const float*)d_in[15];
    const float* c1_bias = (const float*)d_in[16];
    const float* c2_Wl   = (const float*)d_in[17];
    const float* c2_bl   = (const float*)d_in[18];
    const float* c2_Wr   = (const float*)d_in[19];
    const float* c2_br   = (const float*)d_in[20];
    const float* c2_att  = (const float*)d_in[21];
    const float* c2_Wres = (const float*)d_in[22];
    const float* c2_bias = (const float*)d_in[23];
    const float* ln1_g = (const float*)d_in[24];
    const float* ln1_b = (const float*)d_in[25];
    const float* ln2_g = (const float*)d_in[26];
    const float* ln2_b = (const float*)d_in[27];
    const float* h1_W  = (const float*)d_in[28];
    const float* h1_b  = (const float*)d_in[29];
    const float* hln_g = (const float*)d_in[30];
    const float* hln_b = (const float*)d_in[31];
    const float* h2_W  = (const float*)d_in[32];
    const float* h2_b  = (const float*)d_in[33];

    char* wsp = (char*)d_ws;
    auto alloc = [&](size_t bytes) -> char* {
        char* p = wsp;
        wsp += (bytes + 255) & ~(size_t)255;
        return p;
    };
    float*    xl1   = (float*)alloc((size_t)NDATA  * NHC * 4);
    float*    xr1   = (float*)alloc((size_t)NTASKS * NHC * 4);
    float*    xl2   = (float*)alloc((size_t)NTASKS * NHC * 4);
    float*    xr2   = (float*)alloc((size_t)NTASKS * NHC * 4);
    float*    sc1   = (float*)alloc((size_t)NE * 2 * 4);
    float*    sc2   = (float*)alloc((size_t)NE * 2 * 4);
    // zero-init region (contiguous): smax1, smax2, denom1, denom2, acc1, acc2
    unsigned* smax1 = (unsigned*)alloc((size_t)NTASKS * 2 * 4);
    unsigned* smax2 = (unsigned*)alloc((size_t)NTASKS * 2 * 4);
    float*    den1  = (float*)alloc((size_t)NTASKS * 2 * 4);
    float*    den2  = (float*)alloc((size_t)NTASKS * 2 * 4);
    float*    acc1  = (float*)alloc((size_t)NTASKS * NHC * 4);
    float*    acc2  = (float*)alloc((size_t)NTASKS * NHC * 4);
    float*    emb   = (float*)alloc((size_t)NTASKS * EMBD * 4);
    float*    glob  = (float*)alloc((size_t)NB * EMBD * 4);
    _Float16* wp1l  = (_Float16*)alloc(4 * 32 * 16 * 2);
    _Float16* wp1r  = (_Float16*)alloc(4 * 32 * 16 * 2);
    _Float16* wp2l  = (_Float16*)alloc(4 * 32 * 16 * 2);
    _Float16* wp2r  = (_Float16*)alloc(4 * 32 * 16 * 2);

    // 1) zero smax/denom/acc (one contiguous region)
    size_t nz = ((char*)emb - (char*)smax1) / 4;
    k_zero_u32<<<(unsigned)((nz + 255) / 256), 256, 0, stream>>>(smax1, nz);

    // 2) pack weights into WMMA B layout
    k_pack_w<<<4, 256, 0, stream>>>(c1_Wl, DATAD, wp1l);
    k_pack_w<<<4, 256, 0, stream>>>(c1_Wr, TASKD, wp1r);
    k_pack_w<<<4, 256, 0, stream>>>(c2_Wl, TASKD, wp2l);
    k_pack_w<<<4, 256, 0, stream>>>(c2_Wr, TASKD, wp2r);

    // 3) WMMA projections (wave per 16 rows; 256 thr = 128 rows per block)
    k_proj<<<NDATA  / 128, 256, 0, stream>>>(data_x,  NDATA,  DATAD, wp1l, c1_bl, xl1);
    k_proj<<<NTASKS / 128, 256, 0, stream>>>(tasks_x, NTASKS, TASKD, wp1r, c1_br, xr1);
    k_proj<<<NTASKS / 128, 256, 0, stream>>>(tasks_x, NTASKS, TASKD, wp2l, c2_bl, xl2);
    k_proj<<<NTASKS / 128, 256, 0, stream>>>(tasks_x, NTASKS, TASKD, wp2r, c2_br, xr2);

    // 4) edge scores + segment max
    const int* src_dt = ei_dt;        const int* dst_dt = ei_dt + NE;
    const int* src_tt = ei_tt;        const int* dst_tt = ei_tt + NE;
    k_edge_score<true ><<<NE / 256, 256, 0, stream>>>(src_dt, dst_dt, xl1, xr1,
                                                      eattr, c1_We, c1_att, sc1, smax1);
    k_edge_score<false><<<NE / 256, 256, 0, stream>>>(src_tt, dst_tt, xl2, xr2,
                                                      nullptr, nullptr, c2_att, sc2, smax2);

    // 5) exp + denom + weighted scatter-add
    k_edge_accum<<<NE / 256, 256, 0, stream>>>(src_dt, dst_dt, xl1, sc1, smax1, den1, acc1);
    k_edge_accum<<<NE / 256, 256, 0, stream>>>(src_tt, dst_tt, xl2, sc2, smax2, den2, acc2);

    // 6) finalize each conv: normalize, residual, LN, leaky -> emb slices
    k_finalize<<<NTASKS / 256, 256, 0, stream>>>(acc1, den1, tasks_x, c1_Wres, c1_bias,
                                                 ln1_g, ln1_b, emb, TASKD, 1);
    k_finalize<<<NTASKS / 256, 256, 0, stream>>>(acc2, den2, tasks_x, c2_Wres, c2_bias,
                                                 ln2_g, ln2_b, emb, TASKD + NC, 0);

    // 7) per-graph mean pool
    k_pool<<<NB, 128, 0, stream>>>(emb, counts, glob);

    // 8) head MLP -> d_out [128,3]
    k_head<<<1, NB, 0, stream>>>(emb, glob, devices_x, ptr, h1_W, h1_b,
                                 hln_g, hln_b, h2_W, h2_b, (float*)d_out);
}
